// NaiveMerge_33062658244940
// MI455X (gfx1250) — matrix operand order repaired and checked by reading, not compiled, not run on hardware
//
#include <hip/hip_runtime.h>

typedef __attribute__((ext_vector_type(16))) _Float16 v16h;
typedef __attribute__((ext_vector_type(8)))  _Float16 v8h;
typedef __attribute__((ext_vector_type(4)))  _Float16 h4;
typedef __attribute__((ext_vector_type(2)))  _Float16 h2;
typedef __attribute__((ext_vector_type(8)))  float    v8f;
typedef __attribute__((ext_vector_type(4)))  float    f4;

#define SP_D        256   // feature dim (fixed by reference)
#define SP_NROWS    16    // output rows per workgroup (one M tile)
#define SP_KC       64    // nonzeros per chunk (two WMMA K-steps)
#define SP_TPB      256   // 8 waves, 2 N-tiles each
#define SP_LSTRIDE  66    // halves per feature row in LDS (64 + 2 pad = 33 dwords, odd -> conflict-free)

// ---------------------------------------------------------------------------
// Phase 1: m fp32 -> fp16 into workspace. 25.6 MB, stays resident in the
// 192 MB L2, halving all SpMM gather bytes and matching reference numerics.
// ---------------------------------------------------------------------------
__global__ __launch_bounds__(256)
void spmm_cvt_f16(const float* __restrict__ src, _Float16* __restrict__ dst, int n4) {
    int i = blockIdx.x * blockDim.x + threadIdx.x;
    if (i < n4) {
        f4 v = ((const f4*)src)[i];
        h4 h;
        #pragma unroll
        for (int e = 0; e < 4; ++e) h[e] = (_Float16)v[e];
        ((h4*)dst)[i] = h;
    }
}

// ---------------------------------------------------------------------------
// Phase 2: SpMM via WMMA selector-matrix trick.
//   block b -> output rows [16b, 16b+16); binary search row[] for nnz range
//   A (16x32 f16): A[m,k] = vals16[k] iff row[k]==r0+m (selector, branch-free)
//   B (32x16 f16): gathered H16 rows staged feature-major in LDS
//   C += A*B  via v_wmma_f32_16x16x32_f16
//   64 nonzeros per barrier pair; 8 waves x 2 N-tiles x 2 K-halves
// ---------------------------------------------------------------------------
template <bool F16SRC>
__global__ __launch_bounds__(SP_TPB)
void spmm_wmma(const void* __restrict__ msrc,
               const int*  __restrict__ row,
               const int*  __restrict__ col,
               const float* __restrict__ vals,
               float* __restrict__ out,
               int nnz, int N)
{
    __shared__ _Float16 ldsB[SP_D * SP_LSTRIDE];  // feature-major gathered tile (33792 B)
    __shared__ int      ldsRow[SP_KC];
    __shared__ _Float16 ldsVal[SP_KC];

    const int r0   = blockIdx.x * SP_NROWS;
    const int t    = threadIdx.x;
    const int lane = t & 31;
    const int wave = t >> 5;

    const int n_lane = lane & 15;   // B/C column owned by this lane
    const int kg     = lane >> 4;   // lane half-group (K sub-range selector)

    // CSR-like range for rows [r0, r0+16) in sorted row[] (uniform per WG)
    int lo = 0, hi = nnz;
    while (lo < hi) { int mid = (lo + hi) >> 1; if (row[mid] < r0)            lo = mid + 1; else hi = mid; }
    const int start = lo;
    hi = nnz;
    while (lo < hi) { int mid = (lo + hi) >> 1; if (row[mid] < r0 + SP_NROWS) lo = mid + 1; else hi = mid; }
    const int end = lo;

    v8f acc[2] = {v8f{}, v8f{}};     // 2 N-tiles per wave

    // staging assignment: thread owns a quad of k's x 16 features
    const int kq = (t & 15) * 4;     // k = kq..kq+3
    const int fb = (t >> 4) * 16;    // features fb..fb+15

    for (int c0 = start; c0 < end; c0 += SP_KC) {
        const int kcount = (end - c0 < SP_KC) ? (end - c0) : SP_KC;

        // --- stage row ids + fp16 vals (waves 0-1 only; wave-uniform branch) ---
        if (t < SP_KC) {
            int kk = c0 + t; kk = kk < nnz ? kk : nnz - 1;       // clamp, no branch
            ldsRow[t] = (t < kcount) ? row[kk] : -1;             // select kills tail
            ldsVal[t] = (_Float16)vals[kk];
        }

        // --- gather 4 rows x 16 features each; branch-free via index clamp.
        //     Tail k's load a real (finite) row; A's zero column neutralizes it. ---
        if (F16SRC) {
            const _Float16* h16 = (const _Float16*)msrc;
            v8h rv[4][2];
            #pragma unroll
            for (int j = 0; j < 4; ++j) {
                int kk = c0 + kq + j; kk = kk < nnz ? kk : nnz - 1;
                const _Float16* src = h16 + (size_t)col[kk] * SP_D + fb;
                rv[j][0] = *(const v8h*)src;
                rv[j][1] = *(const v8h*)(src + 8);
            }
            #pragma unroll
            for (int e = 0; e < 16; ++e) {                       // pack k-pairs -> b32 stores
                h2 p0, p1;
                p0[0] = rv[0][e >> 3][e & 7]; p0[1] = rv[1][e >> 3][e & 7];
                p1[0] = rv[2][e >> 3][e & 7]; p1[1] = rv[3][e >> 3][e & 7];
                _Float16* dp = &ldsB[(fb + e) * SP_LSTRIDE + kq];
                *(h2*)dp       = p0;
                *(h2*)(dp + 2) = p1;
            }
        } else {
            const float* mf = (const float*)msrc;
            f4 rv[4][4];
            #pragma unroll
            for (int j = 0; j < 4; ++j) {
                int kk = c0 + kq + j; kk = kk < nnz ? kk : nnz - 1;
                const float* src = mf + (size_t)col[kk] * SP_D + fb;
                #pragma unroll
                for (int q = 0; q < 4; ++q) rv[j][q] = *(const f4*)(src + q * 4);
            }
            #pragma unroll
            for (int e = 0; e < 16; ++e) {
                h2 p0, p1;
                p0[0] = (_Float16)rv[0][e >> 2][e & 3]; p0[1] = (_Float16)rv[1][e >> 2][e & 3];
                p1[0] = (_Float16)rv[2][e >> 2][e & 3]; p1[1] = (_Float16)rv[3][e >> 2][e & 3];
                _Float16* dp = &ldsB[(fb + e) * SP_LSTRIDE + kq];
                *(h2*)dp       = p0;
                *(h2*)(dp + 2) = p1;
            }
        }

        // --- speculative prefetch of next chunk's gathered rows (L2-resident) ---
        {
            int nk = c0 + SP_KC + (t & 63); nk = nk < nnz ? nk : nnz - 1;
            const int pc = col[nk];
            if (F16SRC)
                __builtin_prefetch((const _Float16*)msrc + (size_t)pc * SP_D + (t >> 6) * 64, 0, 1);
            else
                __builtin_prefetch((const float*)msrc + (size_t)pc * SP_D + (t >> 6) * 64, 0, 1);
        }
        __syncthreads();

        // --- build A selector fragments (identical in every wave).
        // lanes 0-15:  M=lane,    K in {0..7} U {16..23} (+32 for second half)
        // lanes 16-31: M=lane-16, K in {8..15} U {24..31} (+32) ---
        v16h a[2];
        {
            const int m_row = r0 + n_lane;
            const int kbase = kg * 8;
            #pragma unroll
            for (int h = 0; h < 2; ++h) {
                #pragma unroll
                for (int e = 0; e < 16; ++e) {
                    const int K = h * 32 + ((e < 8) ? (kbase + e) : (16 + kbase + (e - 8)));
                    a[h][e] = (ldsRow[K] == m_row) ? ldsVal[K] : (_Float16)0.0f;
                }
            }
        }

        // --- 2 N-tiles x 2 K-halves per wave ---
        #pragma unroll
        for (int ti = 0; ti < 2; ++ti) {
            const int d0 = wave * 32 + ti * 16;
            #pragma unroll
            for (int kh = 0; kh < 2; ++kh) {
                // lane group kg covers K = kh*32 + {kg*8..kg*8+7} U {16+kg*8..16+kg*8+7} for column n_lane
                // (matches A's table order) in feature-major LDS
                const _Float16* bp = &ldsB[(d0 + n_lane) * SP_LSTRIDE + kh * 32 + kg * 8];
                v16h b;
                #pragma unroll
                for (int e = 0; e < 16; ++e) b[e] = bp[(e < 8) ? e : (e + 8)];
                acc[ti] = __builtin_amdgcn_wmma_f32_16x16x32_f16(
                    false, a[kh], false, b, (short)0, acc[ti], false, false);
            }
        }
        __syncthreads();
    }

    // --- write C tiles: lane = column n_lane, VGPR r -> M = r + kg*8 ---
    #pragma unroll
    for (int ti = 0; ti < 2; ++ti) {
        const int d0 = wave * 32 + ti * 16;
        #pragma unroll
        for (int r = 0; r < 8; ++r) {
            const int mrow = r0 + r + kg * 8;
            if (mrow < N)
                out[(size_t)mrow * SP_D + d0 + n_lane] = acc[ti][r];
        }
    }
}

extern "C" void kernel_launch(void* const* d_in, const int* in_sizes, int n_in,
                              void* d_out, int out_size, void* d_ws, size_t ws_size,
                              hipStream_t stream) {
    const float* m    = (const float*)d_in[0];
    const int*   row  = (const int*)d_in[1];
    const int*   col  = (const int*)d_in[2];
    const float* vals = (const float*)d_in[3];
    float*       out  = (float*)d_out;

    const int ND  = in_sizes[0];        // N * 256
    const int nnz = in_sizes[1];
    const int N   = ND / SP_D;
    const int nblocks = (N + SP_NROWS - 1) / SP_NROWS;

    if (ws_size >= (size_t)ND * sizeof(_Float16)) {
        _Float16* h16 = (_Float16*)d_ws;
        const int n4 = ND / 4;
        spmm_cvt_f16<<<(n4 + 255) / 256, 256, 0, stream>>>(m, h16, n4);
        spmm_wmma<true><<<nblocks, SP_TPB, 0, stream>>>(h16, row, col, vals, out, nnz, N);
    } else {
        spmm_wmma<false><<<nblocks, SP_TPB, 0, stream>>>(m, row, col, vals, out, nnz, N);
    }
}
